// DLRM_20864951124667
// MI455X (gfx1250) — matrix operand order repaired and checked by reading, not compiled, not run on hardware
//
#include <hip/hip_runtime.h>
#include <hip/hip_bf16.h>
#include <stdint.h>

// ---------------------------------------------------------------------------
// DLRM forward, CDNA5 (gfx1250, wave32) WMMA implementation.
//   B=16384, ND=13, F=26, V=131072, D=64, T=27 tokens, INTERACT=415 (pad 416)
//   BOT=(128,64)  TOP=(512,256)
// Matrix ops use v_wmma_f32_16x16x32_f16 (fp16 in, fp32 accumulate).
// ---------------------------------------------------------------------------

#define BDIM   16384
#define ND     13
#define NF     26
#define NV     131072
#define DD     64
#define TT     27
#define TTP    32          // padded token count (rows 27..31 zeroed)
#define NI     415         // 64 + 27*26/2
#define NIP    416         // padded to multiple of 32 for WMMA K
#define TOP1   512
#define TOP2   256

typedef _Float16 half2_t  __attribute__((ext_vector_type(2)));
typedef _Float16 half4_t  __attribute__((ext_vector_type(4)));
typedef _Float16 half8_t  __attribute__((ext_vector_type(8)));
typedef _Float16 v16h     __attribute__((ext_vector_type(16)));
typedef float    f32x4    __attribute__((ext_vector_type(4)));
typedef float    v8f      __attribute__((ext_vector_type(8)));

static __device__ __forceinline__ v16h cat16(half8_t lo, half8_t hi) {
    return __builtin_shufflevector(lo, hi, 0,1,2,3,4,5,6,7,8,9,10,11,12,13,14,15);
}

static __device__ __forceinline__ v8f wmma_f16(v16h a, v16h b, v8f c) {
    // D = A(16x32 f16) * B(32x16 f16) + C(16x16 f32)
    return __builtin_amdgcn_wmma_f32_16x16x32_f16(
        /*neg_a=*/false, a, /*neg_b=*/false, b,
        /*c_mod=*/(short)0, c, /*reuse_a=*/false, /*reuse_b=*/false);
}

// Load A-fragment (16x32, f16) from row-major memory, row stride `ldk` halves.
// ISA 16-bit A layout: lane<16 holds row M=lane, K octets {0-7,16-23};
// lane>=16 holds row M=lane-16, K octets {8-15,24-31}.
static __device__ __forceinline__ v16h load_afrag(const _Float16* base, int ldk,
                                                  int mrow, int hi) {
    const _Float16* p = base + (size_t)mrow * ldk + hi * 8;
    half8_t lo8 = *(const half8_t*)(p);
    half8_t hi8 = *(const half8_t*)(p + 16);
    return cat16(lo8, hi8);
}

// Load B-fragment (32x16, f16) where memory holds B transposed: row n holds
// the K-contiguous values B[k][n]. lane<16: col N=lane, K octets {0-7,16-23};
// lane>=16: col N=lane-16, K octets {8-15,24-31}.
static __device__ __forceinline__ v16h load_bfrag(const _Float16* baseT, int ldk,
                                                  int mrow, int hi) {
    const _Float16* p = baseT + (size_t)mrow * ldk + hi * 8;
    half8_t lo8 = *(const half8_t*)(p);
    half8_t hi8 = *(const half8_t*)(p + 16);
    return cat16(lo8, hi8);
}

// ---------------------------------------------------------------------------
// Kernel A: convert fp32 weight [K,N] -> fp16 transposed [N,Kp] (zero pad K..Kp)
// ---------------------------------------------------------------------------
__global__ __launch_bounds__(256)
void k_convert_wT(const float* __restrict__ src, _Float16* __restrict__ dst,
                  int K, int Kp, int N) {
    int idx = blockIdx.x * 256 + threadIdx.x;
    if (idx >= N * Kp) return;
    int n = idx / Kp;
    int k = idx - n * Kp;
    float v = (k < K) ? src[(size_t)k * N + n] : 0.0f;
    dst[idx] = (_Float16)v;
}

// ---------------------------------------------------------------------------
// Kernel B: bottom MLP. One wave per sample (8 samples / 256-thread block).
//   h = relu(dense @ w1 + b1)  [128];  tok0 = relu(h @ w2 + b2) [64]
// Writes fp16 token row 0 of tok_h[b][32][64] and zeroes pad rows 27..31.
// ---------------------------------------------------------------------------
__global__ __launch_bounds__(256)
void k_bottom_mlp(const float* __restrict__ dense,
                  const float* __restrict__ w1, const float* __restrict__ b1,
                  const float* __restrict__ w2, const float* __restrict__ b2,
                  _Float16* __restrict__ tok_h) {
    __shared__ float hsh[8][128];
    int wave = threadIdx.x >> 5;
    int lane = threadIdx.x & 31;
    int b = blockIdx.x * 8 + wave;

    float dv[ND];
    #pragma unroll
    for (int k = 0; k < ND; ++k) dv[k] = dense[(size_t)b * ND + k];

    #pragma unroll
    for (int i = 0; i < 4; ++i) {
        int j = lane + 32 * i;
        float acc = b1[j];
        #pragma unroll
        for (int k = 0; k < ND; ++k) acc += dv[k] * w1[k * 128 + j];
        hsh[wave][j] = fmaxf(acc, 0.0f);
    }
    // wave-private LDS: DS ops are in-order within a wave, no barrier needed.
    _Float16* row0 = tok_h + (size_t)b * (TTP * DD);
    #pragma unroll
    for (int i = 0; i < 2; ++i) {
        int j = lane + 32 * i;
        float acc = b2[j];
        #pragma unroll
        for (int k = 0; k < 128; ++k) acc += hsh[wave][k] * w2[k * 64 + j];
        row0[j] = (_Float16)fmaxf(acc, 0.0f);
    }
    // zero pad rows 27..31 (5*64 = 320 halves)
    _Float16* pad = row0 + TT * DD;
    for (int p = lane; p < (TTP - TT) * DD; p += 32) pad[p] = (_Float16)0.0f;
}

// ---------------------------------------------------------------------------
// Kernel C: embedding gather. One float4 per thread -> half4 store.
// tok rows 1..26 of tok_h[b][32][64].
// ---------------------------------------------------------------------------
__global__ __launch_bounds__(256)
void k_gather(const int* __restrict__ sparse,
              const float* __restrict__ emb, _Float16* __restrict__ tok_h) {
    int tid = blockIdx.x * 256 + threadIdx.x;          // B*F*16 threads
    int b = tid / (NF * 16);
    int r = tid - b * (NF * 16);
    int f = r >> 4;
    int q = r & 15;
    int row = sparse[(size_t)b * NF + f];
    const f32x4 v = *(const f32x4*)(emb + ((size_t)f * (NV + 1) + row) * DD + q * 4);
    half4_t h = __builtin_convertvector(v, half4_t);
    *(half4_t*)(tok_h + (size_t)b * (TTP * DD) + (1 + f) * DD + q * 4) = h;
}

// ---------------------------------------------------------------------------
// Kernel D: interaction. One wave per sample. X = tok_h[b] (32x64 fp16,
// rows >=27 zero). gram = X*X^T via 3 upper-triangle 16x16 WMMA tiles.
// Emits act0[b][0..63] = dense_tok (fp16 copy), act0[b][64+idx] = dots,
// act0[b][415] = 0 (K pad).
// ---------------------------------------------------------------------------
__global__ __launch_bounds__(256)
void k_interact(const _Float16* __restrict__ tok_h, _Float16* __restrict__ act0) {
    int wave = threadIdx.x >> 5;
    int lane = threadIdx.x & 31;
    int b = blockIdx.x * 8 + wave;
    int mrow = lane & 15, hi = lane >> 4;

    const _Float16* X = tok_h + (size_t)b * (TTP * DD);

    v16h a[2][2], bf[2][2];
    #pragma unroll
    for (int tb = 0; tb < 2; ++tb)
        #pragma unroll
        for (int kc = 0; kc < 2; ++kc) {
            a[tb][kc]  = load_afrag(X + tb * 16 * DD + kc * 32, DD, mrow, hi);
            bf[tb][kc] = load_bfrag(X + tb * 16 * DD + kc * 32, DD, mrow, hi);
        }

    v8f g00 = {0.f,0.f,0.f,0.f,0.f,0.f,0.f,0.f};
    v8f g01 = g00, g11 = g00;
    g00 = wmma_f16(a[0][0], bf[0][0], g00);
    g00 = wmma_f16(a[0][1], bf[0][1], g00);
    g01 = wmma_f16(a[0][0], bf[1][0], g01);
    g01 = wmma_f16(a[0][1], bf[1][1], g01);
    g11 = wmma_f16(a[1][0], bf[1][0], g11);
    g11 = wmma_f16(a[1][1], bf[1][1], g11);

    _Float16* out = act0 + (size_t)b * NIP;
    // dense_tok passthrough (64 fp16) + zero the K-pad slot
    *(half2_t*)(out + lane * 2) = *(const half2_t*)(X + lane * 2);
    if (lane == 0) out[NI] = (_Float16)0.0f;

    // C layout: VGPR v holds (M = v + 8*hi, N = mrow)
    auto emit = [&](int t, int s, float val) {
        if (t < TT && s < TT && t < s) {
            int idx = t * (TT - 1) - (t * (t - 1)) / 2 + (s - t - 1);
            out[DD + idx] = (_Float16)val;
        }
    };
    #pragma unroll
    for (int v = 0; v < 8; ++v) {
        int M = v + hi * 8, N = mrow;
        emit(M,      N,      g00[v]);
        emit(M,      16 + N, g01[v]);
        emit(16 + M, 16 + N, g11[v]);
    }
}

// ---------------------------------------------------------------------------
// Kernels E/F: WMMA GEMM  C[M,N] = relu(A[M,K] * W^T[N,K]^T + bias).
// Block = 256 threads = 8 waves. Block owns MPW*16 rows; each wave owns NPW
// 16-wide N tiles (8*NPW*16 == NTOT) and all MPW M tiles, so each B fragment
// feeds MPW WMMAs. global_prefetch pulls the next K-chunk of the weight
// stream into L1 while the WMMA pipe drains.
// ---------------------------------------------------------------------------
template<int K, int MPW, int NPW, int NTOT>
__global__ __launch_bounds__(256)
void k_gemm_relu(const _Float16* __restrict__ A, const _Float16* __restrict__ WT,
                 const float* __restrict__ bias, _Float16* __restrict__ C) {
    int wave = threadIdx.x >> 5;
    int lane = threadIdx.x & 31;
    int mrow = lane & 15, hi = lane >> 4;
    int m0 = blockIdx.x * (MPW * 16);
    int nb = wave * (NPW * 16);

    v8f acc[MPW][NPW];
    #pragma unroll
    for (int mt = 0; mt < MPW; ++mt)
        #pragma unroll
        for (int t = 0; t < NPW; ++t)
            acc[mt][t] = (v8f){0.f,0.f,0.f,0.f,0.f,0.f,0.f,0.f};

    #pragma unroll 2
    for (int kc = 0; kc < K / 32; ++kc) {
        v16h af[MPW];
        #pragma unroll
        for (int mt = 0; mt < MPW; ++mt)
            af[mt] = load_afrag(A + (size_t)(m0 + mt * 16) * K + kc * 32, K, mrow, hi);
        if (kc + 1 < K / 32) {
            // prefetch next K-chunk of this wave's weight stream (L2 -> L1)
            __builtin_prefetch(WT + (size_t)(nb + mrow) * K + (kc + 1) * 32 + hi * 16, 0, 0);
        }
        #pragma unroll
        for (int t = 0; t < NPW; ++t) {
            v16h bfv = load_bfrag(WT + (size_t)(nb + t * 16) * K + kc * 32, K, mrow, hi);
            #pragma unroll
            for (int mt = 0; mt < MPW; ++mt)
                acc[mt][t] = wmma_f16(af[mt], bfv, acc[mt][t]);
        }
    }
    #pragma unroll
    for (int mt = 0; mt < MPW; ++mt)
        #pragma unroll
        for (int t = 0; t < NPW; ++t) {
            int col = nb + t * 16 + mrow;
            float bb = bias[col];
            #pragma unroll
            for (int v = 0; v < 8; ++v) {
                int row = m0 + mt * 16 + v + hi * 8;
                float val = fmaxf(acc[mt][t][v] + bb, 0.0f);
                C[(size_t)row * NTOT + col] = (_Float16)val;
            }
        }
}

// ---------------------------------------------------------------------------
// Kernel G: logit[b] = dot(act2[b] (256 fp16), w_o (256 f32)) + b_o.
// One wave per row, wave32 shfl_xor reduction.
// ---------------------------------------------------------------------------
__global__ __launch_bounds__(256)
void k_final(const _Float16* __restrict__ act2, const float* __restrict__ w_o,
             const float* __restrict__ b_o, float* __restrict__ out) {
    int wave = threadIdx.x >> 5;
    int lane = threadIdx.x & 31;
    int b = blockIdx.x * 8 + wave;

    half8_t h = *(const half8_t*)(act2 + (size_t)b * TOP2 + lane * 8);
    float acc = 0.0f;
    #pragma unroll
    for (int i = 0; i < 8; ++i) acc += (float)h[i] * w_o[lane * 8 + i];
    #pragma unroll
    for (int off = 16; off >= 1; off >>= 1) acc += __shfl_xor(acc, off, 32);
    if (lane == 0) out[b] = acc + b_o[0];
}

// ---------------------------------------------------------------------------
extern "C" void kernel_launch(void* const* d_in, const int* in_sizes, int n_in,
                              void* d_out, int out_size, void* d_ws, size_t ws_size,
                              hipStream_t stream) {
    (void)in_sizes; (void)n_in; (void)out_size; (void)ws_size;
    const float* dense = (const float*)d_in[0];
    const int*   sparse = (const int*)d_in[1];
    const float* emb  = (const float*)d_in[2];
    const float* w_b1 = (const float*)d_in[3];
    const float* b_b1 = (const float*)d_in[4];
    const float* w_b2 = (const float*)d_in[5];
    const float* b_b2 = (const float*)d_in[6];
    const float* w_t1 = (const float*)d_in[7];
    const float* b_t1 = (const float*)d_in[8];
    const float* w_t2 = (const float*)d_in[9];
    const float* b_t2 = (const float*)d_in[10];
    const float* w_o  = (const float*)d_in[11];
    const float* b_o  = (const float*)d_in[12];
    float* out = (float*)d_out;

    // workspace carve-up (~102 MB)
    char* ws = (char*)d_ws;
    size_t off = 0;
    _Float16* tok_h = (_Float16*)(ws + off); off += (size_t)BDIM * TTP * DD * 2;   // 64 MB
    _Float16* act0  = (_Float16*)(ws + off); off += (size_t)BDIM * NIP * 2;        // 13 MB
    _Float16* act1  = (_Float16*)(ws + off); off += (size_t)BDIM * TOP1 * 2;       // 16 MB
    _Float16* act2  = (_Float16*)(ws + off); off += (size_t)BDIM * TOP2 * 2;       //  8 MB
    _Float16* wT1h  = (_Float16*)(ws + off); off += (size_t)TOP1 * NIP * 2;
    _Float16* wT2h  = (_Float16*)(ws + off); off += (size_t)TOP2 * TOP1 * 2;

    // A) weight conversion + transpose (fp32 [K,N] -> fp16 [N,Kp])
    {
        int n1 = TOP1 * NIP;   // 512*416
        k_convert_wT<<<(n1 + 255) / 256, 256, 0, stream>>>(w_t1, wT1h, NI, NIP, TOP1);
        int n2 = TOP2 * TOP1;  // 256*512
        k_convert_wT<<<(n2 + 255) / 256, 256, 0, stream>>>(w_t2, wT2h, TOP1, TOP1, TOP2);
    }
    // B) bottom MLP -> token row 0 (+ zero pad rows)
    k_bottom_mlp<<<BDIM / 8, 256, 0, stream>>>(dense, w_b1, b_b1, w_b2, b_b2, tok_h);
    // C) embedding gather -> token rows 1..26
    {
        int total = BDIM * NF * 16;
        k_gather<<<total / 256, 256, 0, stream>>>(sparse, emb, tok_h);
    }
    // D) pairwise interaction (WMMA X*X^T) -> act0 [B,416]
    k_interact<<<BDIM / 8, 256, 0, stream>>>(tok_h, act0);
    // E) top MLP 1: [B,416] x [416,512] -> act1   (block = 32 rows x 512 cols)
    k_gemm_relu<NIP, 2, 4, TOP1><<<BDIM / 32, 256, 0, stream>>>(act0, wT1h, b_t1, act1);
    // F) top MLP 2: [B,512] x [512,256] -> act2   (block = 32 rows x 256 cols)
    k_gemm_relu<TOP1, 2, 2, TOP2><<<BDIM / 32, 256, 0, stream>>>(act1, wT2h, b_t2, act2);
    // G) output head
    k_final<<<BDIM / 8, 256, 0, stream>>>(act2, w_o, b_o, out);
}
